// MoETransformerBlock_40827959116456
// MI455X (gfx1250) — matrix operand order repaired and checked by reading, not compiled, not run on hardware
//
// MoE transformer block for MI455X (gfx1250, wave32, WMMA).
// bf16 WMMA (16x16x32, fp32 accum) GEMMs; flash attention with LDS K/V^T
// tiles; sparse top-2 MoE dispatch. Global->LDS staging uses CDNA5
// GLOBAL_LOAD_ASYNC_TO_LDS_B128 (ASYNCcnt) instead of VGPR round-trips.
#include <hip/hip_runtime.h>

typedef __attribute__((ext_vector_type(16))) __bf16 v16bf;
typedef __attribute__((ext_vector_type(8)))  float  v8f;
typedef __attribute__((ext_vector_type(4)))  unsigned int v4u;
typedef __attribute__((ext_vector_type(4)))  float  v4f;

#define TB 256
#define BM 128
#define BN 128
#define BK 32
#define LSTR (BK + 8)   // padded LDS row stride (bf16 elems) to dodge bank conflicts

union FragB16 { v16bf v; v4u q[2]; };

__device__ __forceinline__ v8f zero8() {
  v8f z = {0.f,0.f,0.f,0.f,0.f,0.f,0.f,0.f};
  return z;
}

__device__ __forceinline__ v8f wmma_bf16(const FragB16& a, const FragB16& b, v8f c) {
  return __builtin_amdgcn_wmma_f32_16x16x32_bf16(false, a.v, false, b.v, (short)0, c, false, false);
}

// CDNA5 async copy: 16B global -> LDS, tracked by ASYNCcnt (no VGPR staging).
// LDS address = low 32 bits of the generic pointer (ISA flat->LDS truncation).
__device__ __forceinline__ void async_b128(void* lds, const void* g) {
  const unsigned l = (unsigned)(unsigned long long)lds;
  asm volatile("global_load_async_to_lds_b128 %0, %1, off"
               :: "v"(l), "v"(g) : "memory");
}
__device__ __forceinline__ void async_join() {
  asm volatile("s_wait_asynccnt 0" ::: "memory");
}

// A-operand fragment (16x32): lane half selects K {0..7,16..23} vs {8..15,24..31}
__device__ __forceinline__ void lds_frag_a(FragB16& f, const __bf16* lds, int rowBase, int lane) {
  const __bf16* p = lds + (rowBase + (lane & 15)) * LSTR + ((lane >> 4) << 3);
  f.q[0] = *(const v4u*)p;
  f.q[1] = *(const v4u*)(p + 16);
}

// B-operand fragment (32x16): lane half selects K {0..7,16..23} vs {8..15,24..31}
__device__ __forceinline__ void lds_frag_b(FragB16& f, const __bf16* lds, int rowBase, int lane) {
  const __bf16* p = lds + (rowBase + (lane & 15)) * LSTR + ((lane >> 4) << 3);
  f.q[0] = *(const v4u*)p;
  f.q[1] = *(const v4u*)(p + 16);
}

// Core 128x128 GEMM tile: C = A[M,K] * Bt[N,K]^T. Optional row indirection
// (token gather) and row clamp for ragged MoE tiles.
__device__ __forceinline__ void gemm_core(const __bf16* __restrict__ A,
                                          const int* __restrict__ rowIdx, int rowClamp,
                                          const __bf16* __restrict__ Bt, int Kdim,
                                          int rowBase, int colBase, v8f acc[2][4]) {
  __shared__ __bf16 lA[BM * LSTR];
  __shared__ __bf16 lB[BN * LSTR];
  const int tid = threadIdx.x, lane = tid & 31, w = tid >> 5;
  const int wr = (w >> 1) * 32, wc = (w & 1) * 64;   // 4x2 wave grid, 32x64 per wave
#pragma unroll
  for (int mt = 0; mt < 2; ++mt)
#pragma unroll
    for (int nt = 0; nt < 4; ++nt) acc[mt][nt] = zero8();

  for (int k0 = 0; k0 < Kdim; k0 += BK) {
    for (int s = tid; s < BM * (BK / 8); s += TB) {
      const int r = s >> 2, c = (s & 3) << 3;
      int gr = rowBase + r;
      if (rowClamp > 0 && gr >= rowClamp) gr = rowClamp - 1;
      if (rowIdx) gr = rowIdx[gr];
      const __bf16* gp = A + (size_t)gr * Kdim + k0 + c;
      async_b128(lA + r * LSTR + c, gp);
      const __bf16* gq = Bt + (size_t)(colBase + r) * Kdim + k0 + c;
      async_b128(lB + r * LSTR + c, gq);
      __builtin_prefetch(gq + BK, 0, 3);               // next weight k-tile
    }
    async_join();
    __syncthreads();
    FragB16 fa[2], fb[4];
#pragma unroll
    for (int mt = 0; mt < 2; ++mt) lds_frag_a(fa[mt], lA, wr + mt * 16, lane);
#pragma unroll
    for (int nt = 0; nt < 4; ++nt) lds_frag_b(fb[nt], lB, wc + nt * 16, lane);
#pragma unroll
    for (int mt = 0; mt < 2; ++mt)
#pragma unroll
      for (int nt = 0; nt < 4; ++nt)
        acc[mt][nt] = wmma_bf16(fa[mt], fb[nt], acc[mt][nt]);
    __syncthreads();
  }
}

// C/D layout: element r of lane l -> local row wr+mt*16+((l>>4)<<3)+r, col wc+nt*16+(l&15)
// Variadic so commas in the body don't split macro arguments.
#define GEMM_EPILOGUE(...)                                                    \
  {                                                                           \
    const int lane = threadIdx.x & 31, w = threadIdx.x >> 5;                  \
    const int wr = (w >> 1) * 32, wc = (w & 1) * 64;                          \
    _Pragma("unroll")                                                         \
    for (int mt = 0; mt < 2; ++mt) {                                          \
      _Pragma("unroll")                                                       \
      for (int nt = 0; nt < 4; ++nt) {                                        \
        _Pragma("unroll")                                                     \
        for (int r = 0; r < 8; ++r) {                                         \
          const int lm  = wr + mt * 16 + ((lane >> 4) << 3) + r;              \
          const int ln_ = wc + nt * 16 + (lane & 15);                         \
          const float cv = acc[mt][nt][r];                                    \
          __VA_ARGS__                                                         \
        }                                                                     \
      }                                                                       \
    }                                                                         \
  }

// ---------------- weight convert + transpose: fp32 [R,C] -> bf16 [C,R] -----
__global__ __launch_bounds__(256) void transpose_cvt_kernel(const float* __restrict__ src,
                                                            __bf16* __restrict__ dst,
                                                            int R, int C) {
  src += (size_t)blockIdx.z * R * C;
  dst += (size_t)blockIdx.z * R * C;
  __shared__ float tile[32][33];
  const int tx = threadIdx.x, ty = threadIdx.y;
  const int c0 = blockIdx.x * 32, r0 = blockIdx.y * 32;
#pragma unroll
  for (int i = 0; i < 4; ++i)
    tile[ty + i * 8][tx] = src[(size_t)(r0 + ty + i * 8) * C + c0 + tx];
  __syncthreads();
#pragma unroll
  for (int i = 0; i < 4; ++i)
    dst[(size_t)(c0 + ty + i * 8) * R + r0 + tx] = (__bf16)tile[tx][ty + i * 8];
}

// ---------------- layernorm (one token/block, D=1024), optional fp32 copy --
__global__ __launch_bounds__(256) void ln_kernel(const float* __restrict__ x,
                                                 const float* __restrict__ g,
                                                 const float* __restrict__ b,
                                                 __bf16* __restrict__ outB,
                                                 float* __restrict__ copyOut) {
  const size_t t = blockIdx.x;
  const float* row = x + t * 1024;
  const int tid = threadIdx.x, lane = tid & 31, w = tid >> 5;
  float v[4], s = 0.f, s2 = 0.f;
#pragma unroll
  for (int i = 0; i < 4; ++i) { v[i] = row[tid + i * 256]; s += v[i]; s2 += v[i] * v[i]; }
#pragma unroll
  for (int mask = 16; mask >= 1; mask >>= 1) {
    s  += __shfl_xor(s, mask, 32);
    s2 += __shfl_xor(s2, mask, 32);
  }
  __shared__ float red[8][2];
  if (lane == 0) { red[w][0] = s; red[w][1] = s2; }
  __syncthreads();
  s = 0.f; s2 = 0.f;
#pragma unroll
  for (int i = 0; i < 8; ++i) { s += red[i][0]; s2 += red[i][1]; }
  const float mu = s * (1.f / 1024.f);
  const float var = s2 * (1.f / 1024.f) - mu * mu;
  const float rstd = rsqrtf(var + 1e-5f);
#pragma unroll
  for (int i = 0; i < 4; ++i) {
    const int c = tid + i * 256;
    outB[t * 1024 + c] = (__bf16)((v[i] - mu) * rstd * g[c] + b[c]);
    if (copyOut) copyOut[t * 1024 + c] = v[i];
  }
}

// ---------------- QKV GEMM: n1[T,1024] x WqkvT[3072,1024]; epilogue scatters
// q,k as [B,H,S,64] and v transposed as [B,H,64,S] (WMMA B-operand for P@V).
__global__ __launch_bounds__(TB) void gemm_qkv_kernel(const __bf16* __restrict__ n1,
                                                      const __bf16* __restrict__ wt,
                                                      const float* __restrict__ bias,
                                                      __bf16* __restrict__ q,
                                                      __bf16* __restrict__ k,
                                                      __bf16* __restrict__ vt) {
  v8f acc[2][4];
  gemm_core(n1, nullptr, 0, wt, 1024, blockIdx.y * BM, blockIdx.x * BN, acc);
  GEMM_EPILOGUE(
    const int m = blockIdx.y * BM + lm;
    const int n = blockIdx.x * BN + ln_;
    const float val = cv + bias[n];
    const int b = m >> 10;
    const int s = m & 1023;
    if (n < 1024) {
      q[((size_t)(b * 16 + (n >> 6)) * 1024 + s) * 64 + (n & 63)] = (__bf16)val;
    } else if (n < 2048) {
      const int nn = n - 1024;
      k[((size_t)(b * 16 + (nn >> 6)) * 1024 + s) * 64 + (nn & 63)] = (__bf16)val;
    } else {
      const int nn = n - 2048;
      vt[((size_t)(b * 16 + (nn >> 6)) * 64 + (nn & 63)) * 1024 + s] = (__bf16)val;
    }
  )
}

// ---------------- flash attention: block = 128 q rows of one (b,h); 8 waves,
// 16 q rows each; iterate keys in 64-wide LDS tiles with online softmax.
__global__ __launch_bounds__(TB) void attn_kernel(const __bf16* __restrict__ qg,
                                                  const __bf16* __restrict__ kg,
                                                  const __bf16* __restrict__ vtg,
                                                  __bf16* __restrict__ ao) {
  const int bh = blockIdx.y, b = bh >> 4, h = bh & 15;
  const int q0 = blockIdx.x * 128;
  const int tid = threadIdx.x, lane = tid & 31, w = tid >> 5;
  const __bf16* Q  = qg  + (size_t)bh * 1024 * 64;
  const __bf16* Kp = kg  + (size_t)bh * 1024 * 64;
  const __bf16* Vt = vtg + (size_t)bh * 64 * 1024;

  __shared__ __bf16 lK[64][72];        // [key][d]
  __shared__ __bf16 lV[64][72];        // [d][key]
  __shared__ __bf16 lP[8][16][72];     // per-wave P restage (C-layout -> A-layout)

  FragB16 aq[2];
  {
    const int m = lane & 15, kh = (lane >> 4) << 3;
    const __bf16* qrow = Q + (size_t)(q0 + w * 16 + m) * 64;
    aq[0].q[0] = *(const v4u*)(qrow + kh);
    aq[0].q[1] = *(const v4u*)(qrow + 16 + kh);
    aq[1].q[0] = *(const v4u*)(qrow + 32 + kh);
    aq[1].q[1] = *(const v4u*)(qrow + 48 + kh);
  }
  v8f accO[4];
#pragma unroll
  for (int nt = 0; nt < 4; ++nt) accO[nt] = zero8();
  float mrun[8], lrun[8];
#pragma unroll
  for (int r = 0; r < 8; ++r) { mrun[r] = -1e30f; lrun[r] = 0.f; }

  for (int jb = 0; jb < 1024; jb += 64) {
    __syncthreads();
    for (int s = tid; s < 64 * 8; s += TB) {
      const int r = s >> 3, c = (s & 7) << 3;
      async_b128(&lK[r][c], Kp + (size_t)(jb + r) * 64 + c);
      async_b128(&lV[r][c], Vt + (size_t)r * 1024 + jb + c);
    }
    async_join();
    __syncthreads();
    // S = (Q K^T) / sqrt(64): 4 column tiles of 16 keys
    v8f sacc[4];
#pragma unroll
    for (int jt = 0; jt < 4; ++jt) {
      v8f sv = zero8();
#pragma unroll
      for (int kk = 0; kk < 2; ++kk) {
        FragB16 bk;
        const __bf16* p = &lK[jt * 16 + (lane & 15)][kk * 32 + ((lane >> 4) << 3)];
        bk.q[0] = *(const v4u*)p; bk.q[1] = *(const v4u*)(p + 16);
        sv = wmma_bf16(aq[kk], bk, sv);
      }
      sacc[jt] = sv * 0.125f;
    }
    // per-row max across the 16-lane half (rows ((lane>>4)<<3)+r)
    float pm[8];
#pragma unroll
    for (int r = 0; r < 8; ++r) {
      pm[r] = sacc[0][r];
#pragma unroll
      for (int jt = 1; jt < 4; ++jt) pm[r] = fmaxf(pm[r], sacc[jt][r]);
    }
#pragma unroll
    for (int mask = 1; mask < 16; mask <<= 1)
#pragma unroll
      for (int r = 0; r < 8; ++r) pm[r] = fmaxf(pm[r], __shfl_xor(pm[r], mask, 32));
    float corr[8], ps[8];
#pragma unroll
    for (int r = 0; r < 8; ++r) {
      const float mn = fmaxf(mrun[r], pm[r]);
      corr[r] = __expf(mrun[r] - mn);
      mrun[r] = mn; ps[r] = 0.f;
    }
#pragma unroll
    for (int jt = 0; jt < 4; ++jt)
#pragma unroll
      for (int r = 0; r < 8; ++r) {
        const float p = __expf(sacc[jt][r] - mrun[r]);
        ps[r] += p;
        lP[w][((lane >> 4) << 3) + r][jt * 16 + (lane & 15)] = (__bf16)p;
      }
#pragma unroll
    for (int mask = 1; mask < 16; mask <<= 1)
#pragma unroll
      for (int r = 0; r < 8; ++r) ps[r] += __shfl_xor(ps[r], mask, 32);
#pragma unroll
    for (int r = 0; r < 8; ++r) lrun[r] = lrun[r] * corr[r] + ps[r];
#pragma unroll
    for (int nt = 0; nt < 4; ++nt)
#pragma unroll
      for (int r = 0; r < 8; ++r) accO[nt][r] = accO[nt][r] * corr[r];
    // wave-private LDS region: only need the split DS counter, not a barrier
    asm volatile("s_wait_dscnt 0" ::: "memory");
    // O += P @ V
#pragma unroll
    for (int kk = 0; kk < 2; ++kk) {
      FragB16 ap;
      const __bf16* p = &lP[w][lane & 15][kk * 32 + ((lane >> 4) << 3)];
      ap.q[0] = *(const v4u*)p; ap.q[1] = *(const v4u*)(p + 16);
#pragma unroll
      for (int nt = 0; nt < 4; ++nt) {
        FragB16 bv;
        const __bf16* pv = &lV[nt * 16 + (lane & 15)][kk * 32 + ((lane >> 4) << 3)];
        bv.q[0] = *(const v4u*)pv; bv.q[1] = *(const v4u*)(pv + 16);
        accO[nt] = wmma_bf16(ap, bv, accO[nt]);
      }
    }
  }
#pragma unroll
  for (int nt = 0; nt < 4; ++nt)
#pragma unroll
    for (int r = 0; r < 8; ++r) {
      const int srow = q0 + w * 16 + ((lane >> 4) << 3) + r;
      const int col  = h * 64 + nt * 16 + (lane & 15);
      ao[(size_t)(b * 1024 + srow) * 1024 + col] = (__bf16)(accO[nt][r] / lrun[r]);
    }
}

// ---------------- output projection + residual: x1 = x + ao @ Wo + bo ------
__global__ __launch_bounds__(TB) void gemm_wo_kernel(const __bf16* __restrict__ ao,
                                                     const __bf16* __restrict__ wot,
                                                     const float* __restrict__ bo,
                                                     const float* __restrict__ x,
                                                     float* __restrict__ x1) {
  v8f acc[2][4];
  gemm_core(ao, nullptr, 0, wot, 1024, blockIdx.y * BM, blockIdx.x * BN, acc);
  GEMM_EPILOGUE(
    const size_t m = blockIdx.y * BM + lm;
    const int n = blockIdx.x * BN + ln_;
    x1[m * 1024 + n] = x[m * 1024 + n] + cv + bo[n];
  )
}

// ---------------- router: 1 wave per token; softmax over 8, top-2 dispatch -
__global__ __launch_bounds__(256) void router_kernel(const __bf16* __restrict__ n2,
                                                     const float* __restrict__ Wr,
                                                     float* __restrict__ probSum,
                                                     int* __restrict__ counts,
                                                     int* __restrict__ tokList,
                                                     float* __restrict__ tokGate) {
  const int w = threadIdx.x >> 5, lane = threadIdx.x & 31;
  const int t = blockIdx.x * 8 + w;
  float acc[8];
#pragma unroll
  for (int e = 0; e < 8; ++e) acc[e] = 0.f;
  for (int d = lane; d < 1024; d += 32) {
    const float xv = (float)n2[(size_t)t * 1024 + d];
    const v4f w0 = *(const v4f*)(Wr + d * 8);
    const v4f w1 = *(const v4f*)(Wr + d * 8 + 4);
    acc[0] += xv * w0.x; acc[1] += xv * w0.y; acc[2] += xv * w0.z; acc[3] += xv * w0.w;
    acc[4] += xv * w1.x; acc[5] += xv * w1.y; acc[6] += xv * w1.z; acc[7] += xv * w1.w;
  }
#pragma unroll
  for (int mask = 16; mask >= 1; mask >>= 1)
#pragma unroll
    for (int e = 0; e < 8; ++e) acc[e] += __shfl_xor(acc[e], mask, 32);
  if (lane == 0) {
    float mx = acc[0];
#pragma unroll
    for (int e = 1; e < 8; ++e) mx = fmaxf(mx, acc[e]);
    float p[8], sum = 0.f;
#pragma unroll
    for (int e = 0; e < 8; ++e) { p[e] = __expf(acc[e] - mx); sum += p[e]; }
    const float inv = 1.f / sum;
#pragma unroll
    for (int e = 0; e < 8; ++e) { p[e] *= inv; atomicAdd(&probSum[e], p[e]); }
    int i1 = 0;
#pragma unroll
    for (int e = 1; e < 8; ++e) if (p[e] > p[i1]) i1 = e;
    int i2 = (i1 == 0) ? 1 : 0;
#pragma unroll
    for (int e = 0; e < 8; ++e) if (e != i1 && p[e] > p[i2]) i2 = e;
    const float gs = 1.f / (p[i1] + p[i2]);
    const int s1 = atomicAdd(&counts[i1], 1);
    tokList[i1 * 8192 + s1] = t; tokGate[i1 * 8192 + s1] = p[i1] * gs;
    const int s2 = atomicAdd(&counts[i2], 1);
    tokList[i2 * 8192 + s2] = t; tokGate[i2 * 8192 + s2] = p[i2] * gs;
  }
}

__global__ void init_kernel(int* counts, float* probSum) {
  if (threadIdx.x < 8) { counts[threadIdx.x] = 0; probSum[threadIdx.x] = 0.f; }
}

// prefix offsets for compact h buffer + aux loss
__global__ void finalize_kernel(const int* counts, int* offsets, const float* probSum,
                                float* aux) {
  if (threadIdx.x == 0) {
    int o = 0; float a = 0.f;
    for (int e = 0; e < 8; ++e) {
      offsets[e] = o; o += counts[e];
      a += ((float)counts[e] / (8192.f * 2.f)) * (probSum[e] / 8192.f);
    }
    *aux = 8.f * a;
  }
}

// ---------------- MoE expert up-proj: h = gelu(n2[gather] @ W1[e] + b1[e]) --
__global__ __launch_bounds__(TB) void gemm_moe_up_kernel(const __bf16* __restrict__ n2,
                                                         const __bf16* __restrict__ w1t,
                                                         const float* __restrict__ b1,
                                                         const int* __restrict__ counts,
                                                         const int* __restrict__ offsets,
                                                         const int* __restrict__ tokList,
                                                         __bf16* __restrict__ h) {
  const int e = blockIdx.z;
  const int cnt = counts[e];
  if ((int)(blockIdx.y * BM) >= cnt) return;
  const int off = offsets[e];
  v8f acc[2][4];
  gemm_core(n2, tokList + (size_t)e * 8192, cnt, w1t + (size_t)e * 4096 * 1024, 1024,
            blockIdx.y * BM, blockIdx.x * BN, acc);
  const float* be = b1 + (size_t)e * 4096;
  GEMM_EPILOGUE(
    const int lrow = blockIdx.y * BM + lm;
    if (lrow < cnt) {
      const int n = blockIdx.x * BN + ln_;
      const float val = cv + be[n];
      const float gel = 0.5f * val *
          (1.f + tanhf(0.7978845608f * (val + 0.044715f * val * val * val)));
      h[(size_t)(off + lrow) * 4096 + n] = (__bf16)gel;
    }
  )
}

// ---------------- MoE down-proj + gated scatter-add into output ------------
__global__ __launch_bounds__(TB) void gemm_moe_down_kernel(const __bf16* __restrict__ hB,
                                                           const __bf16* __restrict__ w2t,
                                                           const float* __restrict__ b2,
                                                           const int* __restrict__ counts,
                                                           const int* __restrict__ offsets,
                                                           const int* __restrict__ tokList,
                                                           const float* __restrict__ tokGate,
                                                           float* __restrict__ out) {
  const int e = blockIdx.z;
  const int cnt = counts[e];
  if ((int)(blockIdx.y * BM) >= cnt) return;
  const __bf16* A = hB + (size_t)offsets[e] * 4096;
  v8f acc[2][4];
  gemm_core(A, nullptr, cnt, w2t + (size_t)e * 1024 * 4096, 4096,
            blockIdx.y * BM, blockIdx.x * BN, acc);
  const float* be = b2 + (size_t)e * 1024;
  const int* tl = tokList + (size_t)e * 8192;
  const float* tg = tokGate + (size_t)e * 8192;
  GEMM_EPILOGUE(
    const int lrow = blockIdx.y * BM + lm;
    if (lrow < cnt) {
      const int n = blockIdx.x * BN + ln_;
      atomicAdd(&out[(size_t)tl[lrow] * 1024 + n], tg[lrow] * (cv + be[n]));
    }
  )
}

// ---------------------------------------------------------------------------
extern "C" void kernel_launch(void* const* d_in, const int* in_sizes, int n_in,
                              void* d_out, int out_size, void* d_ws, size_t ws_size,
                              hipStream_t stream) {
  const float* x     = (const float*)d_in[0];
  const float* ln1_g = (const float*)d_in[1];
  const float* ln1_b = (const float*)d_in[2];
  const float* Wqkv  = (const float*)d_in[3];
  const float* bqkv  = (const float*)d_in[4];
  const float* Wo    = (const float*)d_in[5];
  const float* bo    = (const float*)d_in[6];
  const float* ln2_g = (const float*)d_in[7];
  const float* ln2_b = (const float*)d_in[8];
  const float* Wr    = (const float*)d_in[9];
  const float* W1    = (const float*)d_in[10];
  const float* b1    = (const float*)d_in[11];
  const float* W2    = (const float*)d_in[12];
  const float* b2    = (const float*)d_in[13];
  float* out = (float*)d_out;

  const size_t T = 8192, Dm = 1024, DF = 4096, E = 8;
  char* ws = (char*)d_ws;
  size_t o = 0;
  auto take = [&](size_t bytes) -> char* {
    char* p = ws + o;
    o = (o + bytes + 255) & ~(size_t)255;
    return p;
  };
  __bf16* n1    = (__bf16*)take(T * Dm * 2);
  __bf16* wqkvT = (__bf16*)take(3 * Dm * Dm * 2);
  __bf16* qB    = (__bf16*)take(T * Dm * 2);
  __bf16* kB    = (__bf16*)take(T * Dm * 2);
  __bf16* vtB   = (__bf16*)take(T * Dm * 2);
  __bf16* aoB   = (__bf16*)take(T * Dm * 2);
  __bf16* woT   = (__bf16*)take(Dm * Dm * 2);
  float*  x1    = (float*) take(T * Dm * 4);
  __bf16* n2    = (__bf16*)take(T * Dm * 2);
  __bf16* w1T   = (__bf16*)take(E * DF * Dm * 2);
  __bf16* w2T   = (__bf16*)take(E * Dm * DF * 2);
  __bf16* hB    = (__bf16*)take(2 * T * DF * 2);     // exactly K*T routed rows
  int*    counts  = (int*)  take(E * 4);
  int*    offsets = (int*)  take(E * 4);
  float*  probSum = (float*)take(E * 4);
  int*    tokList = (int*)  take(E * T * 4);
  float*  tokGate = (float*)take(E * T * 4);

  init_kernel<<<1, 32, 0, stream>>>(counts, probSum);

  dim3 tb(32, 8);
  transpose_cvt_kernel<<<dim3(3072 / 32, 1024 / 32, 1), tb, 0, stream>>>(Wqkv, wqkvT, 1024, 3072);
  transpose_cvt_kernel<<<dim3(1024 / 32, 1024 / 32, 1), tb, 0, stream>>>(Wo, woT, 1024, 1024);
  transpose_cvt_kernel<<<dim3(4096 / 32, 1024 / 32, 8), tb, 0, stream>>>(W1, w1T, 1024, 4096);
  transpose_cvt_kernel<<<dim3(1024 / 32, 4096 / 32, 8), tb, 0, stream>>>(W2, w2T, 4096, 1024);

  ln_kernel<<<8192, 256, 0, stream>>>(x, ln1_g, ln1_b, n1, nullptr);
  gemm_qkv_kernel<<<dim3(3072 / BN, 8192 / BM), TB, 0, stream>>>(n1, wqkvT, bqkv, qB, kB, vtB);
  attn_kernel<<<dim3(1024 / 128, 128), TB, 0, stream>>>(qB, kB, vtB, aoB);
  gemm_wo_kernel<<<dim3(1024 / BN, 8192 / BM), TB, 0, stream>>>(aoB, woT, bo, x, x1);
  ln_kernel<<<8192, 256, 0, stream>>>(x1, ln2_g, ln2_b, n2, out);   // also out = x1
  router_kernel<<<8192 / 8, 256, 0, stream>>>(n2, Wr, probSum, counts, tokList, tokGate);
  finalize_kernel<<<1, 32, 0, stream>>>(counts, offsets, probSum, out + (size_t)8192 * 1024);
  gemm_moe_up_kernel<<<dim3(4096 / BN, 8192 / BM, 8), TB, 0, stream>>>(
      n2, w1T, b1, counts, offsets, tokList, hB);
  gemm_moe_down_kernel<<<dim3(1024 / BN, 8192 / BM, 8), TB, 0, stream>>>(
      hB, w2T, b2, counts, offsets, tokList, tokGate, out);

  (void)in_sizes; (void)n_in; (void)out_size; (void)ws_size;
}
